// Net_53738630807596
// MI455X (gfx1250) — compile-verified, matrix operand order checked by reading, not run on hardware
//
#include <hip/hip_runtime.h>

typedef _Float16 h16;
typedef __attribute__((ext_vector_type(16))) _Float16 v16h;
typedef __attribute__((ext_vector_type(8)))  _Float16 v8h;
typedef __attribute__((ext_vector_type(8)))  float    v8f;

#define HW_    65536
#define SCALE_ 0.35355339059327373f

__device__ __forceinline__ int fragK(int h, int lane) {
  // K index of half h (0..15) for 16-bit A/B fragments (16x32 / 32x16)
  return (((h >> 1) & 3) << 1) + (h & 1) + ((lane >> 4) << 3) + ((h >> 3) << 4);
}

__device__ __forceinline__ v8f wmma16(v16h a, v16h b, v8f c) {
  return __builtin_amdgcn_wmma_f32_16x16x32_f16(false, a, false, b, (short)0, c,
                                                false, false);
}

// ---------------------------------------------------------------------------
// K0: avgpool(8) + leaky + off/scl 1x1 convs -> float4 grid params per
//     (b*HEADS+h, wy, wx): {offx/32, offy/32, sclx, scly}
// ---------------------------------------------------------------------------
__global__ void k_offscl(const float* __restrict__ x,
                         const float* __restrict__ off_w, const float* __restrict__ off_b,
                         const float* __restrict__ scl_w, const float* __restrict__ scl_b,
                         float4* __restrict__ params) {
  __shared__ float act[64];
  __shared__ float vals[32];
  const int win = blockIdx.x;            // 0..1023
  const int b   = blockIdx.y;            // 0..1
  const int wy  = win >> 5, wx = win & 31;
  const int c   = threadIdx.x;           // 64 threads

  const float* src = x + ((size_t)(b * 64 + c)) * HW_ + (wy * 8) * 256 + wx * 8;
  float s = 0.f;
#pragma unroll
  for (int sy = 0; sy < 8; ++sy)
#pragma unroll
    for (int sx = 0; sx < 8; ++sx)
      s += src[sy * 256 + sx];
  s *= (1.0f / 64.0f);
  act[c] = (s > 0.f) ? s : 0.01f * s;
  __syncthreads();

  if (c < 32) {
    const int o = c & 15;
    const float* w  = (c < 16) ? off_w : scl_w;
    const float* bb = (c < 16) ? off_b : scl_b;
    float acc = bb[o];
#pragma unroll 8
    for (int k = 0; k < 64; ++k) acc += w[o * 64 + k] * act[k];
    vals[c] = acc;
  }
  __syncthreads();

  if (c < 8) {
    const int h = c;
    float4 p;
    p.x = vals[2 * h]     * (1.0f / 32.0f);   // offx / (W/WS)
    p.y = vals[2 * h + 1] * (1.0f / 32.0f);   // offy / (H/WS)
    p.z = vals[16 + 2 * h];                   // sclx
    p.w = vals[16 + 2 * h + 1];               // scly
    params[(b * 8 + h) * 1024 + win] = p;
  }
}

// ---------------------------------------------------------------------------
// K1: QKV GEMM (192x64) x (64 x HW) per (modality, batch), f32 in -> f16 out.
//   * Input tile staged TRANSPOSED in LDS (xT[px][ch], stride 72) so B
//     fragments are two aligned 16B chunks -> ds_load_b128.
//   * M-tiles in two groups of 6 behind a NON-unrolled loop (unroll 1) so the
//     scheduler cannot fuse the groups: ~130 live VGPRs, no scratch spills,
//     dependent WMMAs still 6 apart (> f16 hazard window, no v_nop).
//   * Outputs go to purpose-built layouts, one b128 store per M-tile per lane:
//       qwin: (mod,b,head,win,query,hd)   kwin/vwin: (mod,b,head,pixel,hd)
// ---------------------------------------------------------------------------
__global__ void k_qkv(const float* __restrict__ x, const float* __restrict__ lms,
                      const float* __restrict__ qkv_w, const float* __restrict__ qkv_b,
                      h16* __restrict__ qwin, h16* __restrict__ kwin,
                      h16* __restrict__ vwin) {
  __shared__ h16 wlds[192 * 64];
  __shared__ h16 xT[64 * 72];
  const int mod = blockIdx.y;
  const int b   = blockIdx.x >> 10;           // HW/64 = 1024 tiles per batch
  const int p0  = (blockIdx.x & 1023) * 64;
  const float* src = (mod == 0 ? x : lms) + (size_t)b * 64 * HW_;
  const int t = threadIdx.x;                  // 128 threads, 4 waves

  __builtin_prefetch(qkv_w + t * 96, 0, 1);   // warm weight panel (global_prefetch_b8)
  for (int i = t; i < 192 * 64; i += 128) wlds[i] = (h16)qkv_w[i];
  for (int i = t; i < 64 * 64; i += 128) {
    const int ch = i >> 6, px = i & 63;       // consecutive t -> consecutive px (coalesced)
    xT[px * 72 + ch] = (h16)src[(size_t)ch * HW_ + p0 + px];
  }
  __syncthreads();

  const int wave = t >> 5, lane = t & 31;
  const int pxl = (wave << 4) + (lane & 15);  // pixel column within 64-tile

  v16h bfr[2];
#pragma unroll
  for (int ks = 0; ks < 2; ++ks)
#pragma unroll
    for (int h = 0; h < 16; ++h)
      bfr[ks][h] = xT[pxl * 72 + fragK(h, lane) + ks * 32];

  const int mlane = lane & 15;
  const int mrow = (lane >> 4) << 3;

  const int p = p0 + pxl;
  const int yy = p >> 8, xx = p & 255;
  const int win   = (yy >> 3) * 32 + (xx >> 3);
  const int query = (yy & 7) * 8 + (xx & 7);
  const size_t mb = (size_t)(mod * 2 + b);

#pragma unroll 1
  for (int g = 0; g < 2; ++g) {               // two groups of 6 M-tiles (rolled!)
    v8f acc[6];
#pragma unroll
    for (int m6 = 0; m6 < 6; ++m6) acc[m6] = {};

#pragma unroll
    for (int ks = 0; ks < 2; ++ks) {
#pragma unroll
      for (int m6 = 0; m6 < 6; ++m6) {
        const int mt = g * 6 + m6;
        v16h afr;
#pragma unroll
        for (int h = 0; h < 16; ++h)
          afr[h] = wlds[(mt * 16 + mlane) * 64 + fragK(h, lane) + ks * 32];
        acc[m6] = wmma16(afr, bfr[ks], acc[m6]);
      }
    }

#pragma unroll
    for (int m6 = 0; m6 < 6; ++m6) {
      const int mt = g * 6 + m6;
      const int head = ((mt & 3) << 1) + (mrow >> 3);
      v8h vv;
#pragma unroll
      for (int i = 0; i < 8; ++i)
        vv[i] = (h16)(acc[m6][i] + qkv_b[mt * 16 + mrow + i]);
      if (mt < 4)
        *(v8h*)(qwin + ((mb * 8 + head) * 1024 + win) * 512 + (size_t)query * 8) = vv;
      else if (mt < 8)
        *(v8h*)(kwin + (mb * 8 + head) * ((size_t)HW_ * 8) + (size_t)p * 8) = vv;
      else
        *(v8h*)(vwin + (mb * 8 + head) * ((size_t)HW_ * 8) + (size_t)p * 8) = vv;
    }
  }
}

// ---------------------------------------------------------------------------
// K2: fused deformable sampling + window attention.
//     block = (window, head, branch*2+b), 128 threads (4 waves x 16 queries)
//     Sampling: hf=0 thread does K, hf=1 does V; each tap is one b128 load.
// ---------------------------------------------------------------------------
__global__ void k_attn(const h16* __restrict__ qwin,
                       const h16* __restrict__ kwin, const h16* __restrict__ vwin,
                       const float4* __restrict__ params,
                       const float* __restrict__ rpb,
                       h16* __restrict__ attn_out) {
  __shared__ h16 kS[64 * 8];       // [key][hd]  (QK^T B-fragments: hd contiguous)
  __shared__ h16 vT[8 * 72];       // [hd][key]  (attnV B-fragments: key contiguous)
  __shared__ h16 probs[4 * 16 * 64];
  __shared__ float rpbS[225];

  const int win = blockIdx.x, wy = win >> 5, wx = win & 31;
  const int hh  = blockIdx.y;
  const int z   = blockIdx.z;          // branch*2 + b
  const int branch = z >> 1, b = z & 1;
  const int qmod  = (branch == 0) ? 1 : 0;   // branch0: lms queries on pan K/V
  const int kvmod = 1 - qmod;

  const h16* qbase = qwin + (((size_t)(qmod * 2 + b) * 8 + hh) * 1024 + win) * 512;
  const size_t kvoff = ((size_t)(kvmod * 2 + b) * 8 + hh) * ((size_t)HW_ * 8);
  const float4 prm = params[(b * 8 + hh) * 1024 + win];

  const int t = threadIdx.x;
  for (int i = t; i < 225; i += 128) rpbS[i] = rpb[i * 8 + hh];

  { // ---- bilinear sampling: 64 positions, thread pair (K | V) ----
    const int pos = t >> 1, hf = t & 1;
    const int ay = pos >> 3, ax = pos & 7;
    const float cx = (-1.f + (2.f / 255.f) * (wx * 8 + ax))
                   + ((2.f * ax - 7.f) / 255.f) * prm.z + prm.x;
    const float cy = (-1.f + (2.f / 255.f) * (wy * 8 + ay))
                   + ((2.f * ay - 7.f) / 255.f) * prm.w + prm.y;
    const float px = (cx + 1.f) * 0.5f * 255.f;
    const float py = (cy + 1.f) * 0.5f * 255.f;
    const float fx = floorf(px), fy = floorf(py);
    const int x0 = (int)fx, y0 = (int)fy;
    const float wx1 = px - fx, wy1 = py - fy;
    const float wx0 = 1.f - wx1, wy0 = 1.f - wy1;
    const bool vx0 = (x0 >= 0) & (x0 <= 255), vx1 = (x0 + 1 >= 0) & (x0 + 1 <= 255);
    const bool vy0 = (y0 >= 0) & (y0 <= 255), vy1 = (y0 + 1 >= 0) & (y0 + 1 <= 255);
    const int xc0 = min(max(x0, 0), 255),     xc1 = min(max(x0 + 1, 0), 255);
    const int yc0 = min(max(y0, 0), 255),     yc1 = min(max(y0 + 1, 0), 255);
    const float w00 = vx0 && vy0 ? wx0 * wy0 : 0.f;
    const float w10 = vx1 && vy0 ? wx1 * wy0 : 0.f;
    const float w01 = vx0 && vy1 ? wx0 * wy1 : 0.f;
    const float w11 = vx1 && vy1 ? wx1 * wy1 : 0.f;
    const h16* m = (hf ? vwin : kwin) + kvoff;
    const v8h t00 = *(const v8h*)(m + (size_t)(yc0 * 256 + xc0) * 8);
    const v8h t10 = *(const v8h*)(m + (size_t)(yc0 * 256 + xc1) * 8);
    const v8h t01 = *(const v8h*)(m + (size_t)(yc1 * 256 + xc0) * 8);
    const v8h t11 = *(const v8h*)(m + (size_t)(yc1 * 256 + xc1) * 8);
    if (hf == 0) {
      v8h r;
#pragma unroll
      for (int ch = 0; ch < 8; ++ch)
        r[ch] = (h16)(w00 * (float)t00[ch] + w10 * (float)t10[ch]
                    + w01 * (float)t01[ch] + w11 * (float)t11[ch]);
      *(v8h*)(kS + pos * 8) = r;
    } else {
#pragma unroll
      for (int ch = 0; ch < 8; ++ch)
        vT[ch * 72 + pos] = (h16)(w00 * (float)t00[ch] + w10 * (float)t10[ch]
                                + w01 * (float)t01[ch] + w11 * (float)t11[ch]);
    }
  }
  __syncthreads();

  const int wave = t >> 5, lane = t & 31;
  const int qb = wave * 16;

  // ---- Q A-fragment: one b128 global load per lane (K padded 8->32) ----
  v16h qa;
#pragma unroll
  for (int h = 0; h < 16; ++h) qa[h] = (h16)0.f;
  if (lane < 16) {
    const v8h qv = *(const v8h*)(qbase + (size_t)(qb + lane) * 8);
#pragma unroll
    for (int h = 0; h < 8; ++h) qa[h] = qv[h];
  }

  // ---- Q.K^T: preload 4 key fragments (b128 each), 4 independent WMMAs ----
  v16h kb[4];
#pragma unroll
  for (int tt = 0; tt < 4; ++tt) {
#pragma unroll
    for (int h = 0; h < 16; ++h) kb[tt][h] = (h16)0.f;
    if (lane < 16) {
      const v8h kv = *(const v8h*)(kS + (tt * 16 + lane) * 8);
#pragma unroll
      for (int h = 0; h < 8; ++h) kb[tt][h] = kv[h];
    }
  }
  v8f c[4];
#pragma unroll
  for (int tt = 0; tt < 4; ++tt) c[tt] = {};
#pragma unroll
  for (int tt = 0; tt < 4; ++tt) c[tt] = wmma16(qa, kb[tt], c[tt]);

  // ---- scale + relative-position bias (LDS) + softmax over 64 keys ----
  const int mrow = (lane >> 4) << 3;
  const int ncol = lane & 15;
  float sv[4][8];
#pragma unroll
  for (int tt = 0; tt < 4; ++tt) {
#pragma unroll
    for (int i = 0; i < 8; ++i) {
      const int q = qb + mrow + i;
      const int key = tt * 16 + ncol;
      const int dy = (q >> 3) - (key >> 3) + 7;
      const int dx = (q & 7) - (key & 7) + 7;
      sv[tt][i] = c[tt][i] * SCALE_ + rpbS[dy * 15 + dx];
    }
  }
#pragma unroll
  for (int i = 0; i < 8; ++i) {
    float mx = fmaxf(fmaxf(sv[0][i], sv[1][i]), fmaxf(sv[2][i], sv[3][i]));
#pragma unroll
    for (int msk = 1; msk < 16; msk <<= 1) mx = fmaxf(mx, __shfl_xor(mx, msk, 32));
    float e[4], sum = 0.f;
#pragma unroll
    for (int tt = 0; tt < 4; ++tt) { e[tt] = __expf(sv[tt][i] - mx); sum += e[tt]; }
#pragma unroll
    for (int msk = 1; msk < 16; msk <<= 1) sum += __shfl_xor(sum, msk, 32);
    const float inv = 1.f / sum;
#pragma unroll
    for (int tt = 0; tt < 4; ++tt)
      probs[wave * 1024 + (mrow + i) * 64 + tt * 16 + ncol] = (h16)(e[tt] * inv);
  }
  __syncthreads();

  // ---- attn @ V: two independent zero-seeded WMMAs, combine after ----
  v16h pa0, pa1, vb0, vb1;
  {
    const int m = lane & 15;
#pragma unroll
    for (int h = 0; h < 16; ++h) {
      pa0[h] = probs[wave * 1024 + m * 64 + fragK(h, lane)];
      pa1[h] = probs[wave * 1024 + m * 64 + fragK(h, lane) + 32];
    }
#pragma unroll
    for (int h = 0; h < 16; ++h) { vb0[h] = (h16)0.f; vb1[h] = (h16)0.f; }
    if ((lane & 15) < 8) {
      const int n = lane & 15;
#pragma unroll
      for (int h = 0; h < 16; ++h) {
        vb0[h] = vT[n * 72 + fragK(h, lane)];
        vb1[h] = vT[n * 72 + fragK(h, lane) + 32];
      }
    }
  }
  v8f d0 = {}, d1 = {};
  d0 = wmma16(pa0, vb0, d0);
  d1 = wmma16(pa1, vb1, d1);

  // ---- store: 8 accumulator elements = 8 consecutive x -> one b128 ----
  const int hd = lane & 15;
  if (hd < 8) {
    const size_t obase = ((size_t)(branch * 2 + b) * 64 + hh * 8 + hd) * HW_;
    const int q0 = qb + mrow;                       // multiple of 8
    const int y = wy * 8 + (q0 >> 3);
    v8h r;
#pragma unroll
    for (int i = 0; i < 8; ++i) r[i] = (h16)(d0[i] + d1[i]);
    *(v8h*)(attn_out + obase + (size_t)y * 256 + wx * 8) = r;
  }
}

// ---------------------------------------------------------------------------
// K3: proj GEMM (64x64) x (64 x HW) per (branch, batch) + bias -> fp32 out.
//     Input tile staged transposed in LDS so B fragments are ds_load_b128.
// ---------------------------------------------------------------------------
__global__ void k_proj(const h16* __restrict__ attn_out,
                       const float* __restrict__ proj_w, const float* __restrict__ proj_b,
                       float* __restrict__ out) {
  __shared__ h16 wlds[64 * 64];
  __shared__ h16 xT[64 * 72];
  const int branch = blockIdx.y;
  const int b  = blockIdx.x >> 10;
  const int p0 = (blockIdx.x & 1023) * 64;
  const h16* src = attn_out + (size_t)(branch * 2 + b) * 64 * HW_;
  const int t = threadIdx.x;

  __builtin_prefetch(proj_w + t * 32, 0, 1);  // warm weight panel
  for (int i = t; i < 4096; i += 128) wlds[i] = (h16)proj_w[i];
  for (int i = t; i < 4096; i += 128) {
    const int ch = i >> 6, px = i & 63;       // coalesced b16 reads per channel row
    xT[px * 72 + ch] = src[(size_t)ch * HW_ + p0 + px];
  }
  __syncthreads();

  const int wave = t >> 5, lane = t & 31;
  const int pxl = (wave << 4) + (lane & 15);

  v16h bfr[2];
#pragma unroll
  for (int ks = 0; ks < 2; ++ks)
#pragma unroll
    for (int h = 0; h < 16; ++h)
      bfr[ks][h] = xT[pxl * 72 + fragK(h, lane) + ks * 32];

  const int mlane = lane & 15;
  const int mrow = (lane >> 4) << 3;

  v8f acc[4];
#pragma unroll
  for (int mt = 0; mt < 4; ++mt) acc[mt] = {};

#pragma unroll
  for (int ks = 0; ks < 2; ++ks) {
#pragma unroll
    for (int mt = 0; mt < 4; ++mt) {
      v16h afr;
#pragma unroll
      for (int h = 0; h < 16; ++h)
        afr[h] = wlds[(mt * 16 + mlane) * 64 + fragK(h, lane) + ks * 32];
      acc[mt] = wmma16(afr, bfr[ks], acc[mt]);
    }
  }

  const size_t obase = (size_t)(branch * 2 + b) * 64 * HW_;
  const int p = p0 + pxl;
#pragma unroll
  for (int mt = 0; mt < 4; ++mt) {
#pragma unroll
    for (int i = 0; i < 8; ++i) {
      const int o = mt * 16 + mrow + i;
      out[obase + (size_t)o * HW_ + p] = acc[mt][i] + proj_b[o];
    }
  }
}

// ---------------------------------------------------------------------------
extern "C" void kernel_launch(void* const* d_in, const int* in_sizes, int n_in,
                              void* d_out, int out_size, void* d_ws, size_t ws_size,
                              hipStream_t stream) {
  const float* x      = (const float*)d_in[0];
  const float* lms    = (const float*)d_in[1];
  const float* qkv_w  = (const float*)d_in[2];
  const float* qkv_b  = (const float*)d_in[3];
  const float* proj_w = (const float*)d_in[4];
  const float* proj_b = (const float*)d_in[5];
  const float* off_w  = (const float*)d_in[6];
  const float* off_b  = (const float*)d_in[7];
  const float* scl_w  = (const float*)d_in[8];
  const float* scl_b  = (const float*)d_in[9];
  const float* rpb    = (const float*)d_in[10];
  float* out = (float*)d_out;

  // workspace carve-up
  char* ws = (char*)d_ws;
  float4* params = (float4*)ws;                            //    256 KB
  h16* qwin   = (h16*)(ws + 262144ull);                    //     32 MB
  h16* kwin   = (h16*)(ws + 262144ull + 33554432ull);      //     32 MB
  h16* vwin   = (h16*)(ws + 262144ull + 67108864ull);      //     32 MB
  h16* attn_o = (h16*)(ws + 262144ull + 100663296ull);     //     32 MB

  k_offscl<<<dim3(1024, 2), 64, 0, stream>>>(x, off_w, off_b, scl_w, scl_b, params);
  k_qkv   <<<dim3(2048, 2), 128, 0, stream>>>(x, lms, qkv_w, qkv_b, qwin, kwin, vwin);
  k_attn  <<<dim3(1024, 8, 4), 128, 0, stream>>>(qwin, kwin, vwin, params, rpb, attn_o);
  k_proj  <<<dim3(2048, 2), 128, 0, stream>>>(attn_o, proj_w, proj_b, out);
}
